// MultiHeadCrossAttention_91250875171618
// MI455X (gfx1250) — matrix operand order repaired and checked by reading, not compiled, not run on hardware
//
#include <hip/hip_runtime.h>

typedef __attribute__((ext_vector_type(16))) __bf16 v16bf;
typedef __attribute__((ext_vector_type(8)))  __bf16 v8bf;
typedef __attribute__((ext_vector_type(8)))  float  v8f;
typedef unsigned int u32x4 __attribute__((ext_vector_type(4)));
typedef int          i32x8 __attribute__((ext_vector_type(8)));
typedef int          i32x4 __attribute__((ext_vector_type(4)));
typedef __attribute__((address_space(3))) const void* lds_cptr;

#define C_DIM 1024
#define T_DIM 2048
#define S_DIM 2048
#define HD    64
#define NHEAD 16
#define BATCH 2

// ---------------- WMMA helpers (CDNA5 layouts) ----------------

// A fragment: 16x32 bf16, row-major source.
__device__ __forceinline__ v16bf load_a_frag(const __bf16* base, int lda) {
  int lane = threadIdx.x & 31;
  int m = lane & 15, hf = lane >> 4;
  const __bf16* p = base + (long)m * lda + hf * 8;
  v8bf lo = *reinterpret_cast<const v8bf*>(p);
  v8bf hi = *reinterpret_cast<const v8bf*>(p + 16);
  return __builtin_shufflevector(lo, hi, 0,1,2,3,4,5,6,7,8,9,10,11,12,13,14,15);
}

// B fragment: 32x16 from row-major Bt[N][K] (Bt[n][k] == B[k][n]).
__device__ __forceinline__ v16bf load_b_frag(const __bf16* base, int ldb) {
  int lane = threadIdx.x & 31;
  int n = lane & 15, hf = lane >> 4;
  const __bf16* p = base + (long)n * ldb + hf * 8;
  v8bf lo = *reinterpret_cast<const v8bf*>(p);
  v8bf hi = *reinterpret_cast<const v8bf*>(p + 16);
  return __builtin_shufflevector(lo, hi, 0,1,2,3,4,5,6,7,8,9,10,11,12,13,14,15);
}

__device__ __forceinline__ v8f wmma_bf16(v16bf a, v16bf b, v8f c) {
  return __builtin_amdgcn_wmma_f32_16x16x32_bf16(false, a, false, b, (short)0, c, false, false);
}

// Row reduction over the 16 lanes of each half-wave (C/D layout: lane = column).
__device__ __forceinline__ float row_reduce_max(float v) {
  v = fmaxf(v, __int_as_float(__builtin_amdgcn_ds_swizzle(__float_as_int(v), 0x041F)));
  v = fmaxf(v, __int_as_float(__builtin_amdgcn_ds_swizzle(__float_as_int(v), 0x081F)));
  v = fmaxf(v, __int_as_float(__builtin_amdgcn_ds_swizzle(__float_as_int(v), 0x101F)));
  v = fmaxf(v, __int_as_float(__builtin_amdgcn_ds_swizzle(__float_as_int(v), 0x201F)));
  return v;
}
__device__ __forceinline__ float row_reduce_sum(float v) {
  v += __int_as_float(__builtin_amdgcn_ds_swizzle(__float_as_int(v), 0x041F));
  v += __int_as_float(__builtin_amdgcn_ds_swizzle(__float_as_int(v), 0x081F));
  v += __int_as_float(__builtin_amdgcn_ds_swizzle(__float_as_int(v), 0x101F));
  v += __int_as_float(__builtin_amdgcn_ds_swizzle(__float_as_int(v), 0x201F));
  return v;
}

// ---------------- TDM staging: 64x1024 bf16 weight slice -> LDS ----------------
__device__ __forceinline__ void stage_b_slice(const __bf16* gsrc, __bf16* Bs) {
#if defined(__has_builtin) && __has_builtin(__builtin_amdgcn_tensor_load_to_lds)
  if (threadIdx.x < 32) {              // wave 0 only (uniform branch)
    unsigned long long ga = (unsigned long long)gsrc;
    unsigned ldsoff = (unsigned)(unsigned long long)(lds_cptr)Bs;
    u32x4 g0;
    g0[0] = 1u;                                         // count=1, load descriptor
    g0[1] = ldsoff;                                     // lds_addr (bytes)
    g0[2] = (unsigned)(ga & 0xFFFFFFFFu);               // global_addr[31:0]
    g0[3] = (unsigned)((ga >> 32) & 0x1FFFFFFu) | (2u << 30);  // addr[56:32] | type=2
    i32x8 g1;
    g1[0] = 0x00010000;              // workgroup_mask=0, data_size=1 (2 bytes)
    g1[1] = (int)(1024u << 16);      // tensor_dim0 = 1024 (low 16 bits @ bit48)
    g1[2] = (int)(64u << 16);        // tensor_dim0 hi=0 | tensor_dim1 = 64
    g1[3] = (int)(1024u << 16);      // tensor_dim1 hi=0 | tile_dim0 = 1024
    g1[4] = 64;                      // tile_dim1 = 64, tile_dim2 = 0
    g1[5] = 1024;                    // tensor_dim0_stride = 1024
    g1[6] = 0;
    g1[7] = 0;
    i32x4 z4 = {0, 0, 0, 0};
#if __clang_major__ >= 23
    i32x8 z8 = {0, 0, 0, 0, 0, 0, 0, 0};
    __builtin_amdgcn_tensor_load_to_lds(g0, g1, z4, z4, z8, 0);
#else
    __builtin_amdgcn_tensor_load_to_lds(g0, g1, z4, z4, 0);
#endif
    __builtin_amdgcn_s_wait_tensorcnt((short)0);
  }
#else
  const v8bf* src = (const v8bf*)gsrc;
  v8bf* dst = (v8bf*)Bs;
  for (int i = threadIdx.x; i < 64 * C_DIM / 8; i += 256) dst[i] = src[i];
#endif
  __syncthreads();
}

__device__ __forceinline__ v16bf lds_b_frag(const __bf16* Bs, int f, int ln, int hf, int k0) {
  const __bf16* p = &Bs[(f * 16 + ln) * C_DIM + k0 + hf * 8];
  v8bf lo = *(const v8bf*)p;
  v8bf hi = *(const v8bf*)(p + 16);
  return __builtin_shufflevector(lo, hi, 0,1,2,3,4,5,6,7,8,9,10,11,12,13,14,15);
}

// K-loop core, x2-unrolled ping-pong: no register copies, loads for one
// half-step overlap the WMMAs of the other.
__device__ __forceinline__ void gemm_core(const __bf16* a_base, const __bf16* Bs, v8f acc[4]) {
  int lane = threadIdx.x & 31;
  int ln = lane & 15, hf = lane >> 4;
  v16bf a0 = load_a_frag(a_base, C_DIM);
  v16bf b0[4];
#pragma unroll
  for (int f = 0; f < 4; ++f) b0[f] = lds_b_frag(Bs, f, ln, hf, 0);
  for (int k0 = 0; k0 < C_DIM; k0 += 64) {
    v16bf a1 = load_a_frag(a_base + k0 + 32, C_DIM);
    v16bf b1[4];
#pragma unroll
    for (int f = 0; f < 4; ++f) b1[f] = lds_b_frag(Bs, f, ln, hf, k0 + 32);
#pragma unroll
    for (int f = 0; f < 4; ++f) acc[f] = wmma_bf16(a0, b0[f], acc[f]);
    if (k0 + 64 < C_DIM) {
      a0 = load_a_frag(a_base + k0 + 64, C_DIM);
#pragma unroll
      for (int f = 0; f < 4; ++f) b0[f] = lds_b_frag(Bs, f, ln, hf, k0 + 64);
    }
#pragma unroll
    for (int f = 0; f < 4; ++f) acc[f] = wmma_bf16(a1, b1[f], acc[f]);
  }
}

// ---------------- conversion / re-layout kernels ----------------

__global__ void cvt_f32_bf16_kernel(const float* __restrict__ in, __bf16* __restrict__ out, int n) {
  int i = blockIdx.x * blockDim.x + threadIdx.x;
  if (i < n) out[i] = (__bf16)in[i];
}

// W: [H=16, C=1024, hd=64] fp32 -> Wt: [N=1024][K=1024] bf16 with N = h*64+d
__global__ void wqkv_transpose_kernel(const float* __restrict__ W, __bf16* __restrict__ Wt) {
  int i = blockIdx.x * blockDim.x + threadIdx.x;   // over 1M
  int c = i & 1023, n = i >> 10;
  int h = n >> 6, d = n & 63;
  Wt[i] = (__bf16)W[(((long)h << 10) + c) * 64 + d];
}

// ---------------- QKV projection GEMM ----------------
// block = 128 rows x 64 cols (8 waves x 16 rows); B slice staged via TDM.
// mode 0/1: store [BH, T, 64]; mode 2: store transposed [BH, 64, S].
__global__ __launch_bounds__(256) void gemm_qkv_kernel(const __bf16* __restrict__ X,
                                                       const __bf16* __restrict__ Wt,
                                                       const float* __restrict__ bias,
                                                       __bf16* __restrict__ out, int mode) {
  __shared__ __align__(32) __bf16 Bs[64 * C_DIM];   // 128 KB
  int nBlock = blockIdx.x & 15;
  int mBlock = blockIdx.x >> 4;
  int n0 = nBlock * 64;
  stage_b_slice(Wt + (long)n0 * C_DIM, Bs);

  int wave = threadIdx.x >> 5;
  int m0 = mBlock * 128 + wave * 16;
  int lane = threadIdx.x & 31;
  int ln = lane & 15, hf = lane >> 4;

  v8f acc[4] = {};
  gemm_core(X + (long)m0 * C_DIM, Bs, acc);

#pragma unroll
  for (int f = 0; f < 4; ++f) {
    int n = n0 + f * 16 + ln;            // n = h*64 + d
    int h = n >> 6, d = n & 63;
    float bv = bias[n];
#pragma unroll
    for (int j = 0; j < 8; ++j) {
      int row = m0 + j + 8 * hf;         // row = b*2048 + t
      int bb = row >> 11, ts = row & 2047;
      float val = acc[f][j] + bv;
      long idx;
      if (mode == 2)
        idx = ((long)(bb * NHEAD + h) * HD + d) * S_DIM + ts;     // vT
      else
        idx = ((long)(bb * NHEAD + h) * T_DIM + ts) * HD + d;     // q / k
      out[idx] = (__bf16)val;
    }
  }
}

// ---------------- output projection GEMM (f32 out) ----------------
__global__ __launch_bounds__(256) void gemm_out_kernel(const __bf16* __restrict__ A,
                                                       const __bf16* __restrict__ Wt,
                                                       float* __restrict__ Y) {
  __shared__ __align__(32) __bf16 Bs[64 * C_DIM];   // 128 KB
  int nBlock = blockIdx.x & 15;
  int mBlock = blockIdx.x >> 4;
  int n0 = nBlock * 64;
  stage_b_slice(Wt + (long)n0 * C_DIM, Bs);

  int wave = threadIdx.x >> 5;
  int m0 = mBlock * 128 + wave * 16;
  int lane = threadIdx.x & 31;
  int ln = lane & 15, hf = lane >> 4;

  v8f acc[4] = {};
  gemm_core(A + (long)m0 * C_DIM, Bs, acc);

#pragma unroll
  for (int f = 0; f < 4; ++f) {
    int n = n0 + f * 16 + ln;
#pragma unroll
    for (int j = 0; j < 8; ++j) {
      int row = m0 + j + 8 * hf;
      Y[(long)row * C_DIM + n] = acc[f][j];
    }
  }
}

// ---------------- flash attention ----------------

__device__ __forceinline__ void load_kfrags(v16bf kf[4], const __bf16* K, int bh, int s0) {
#pragma unroll
  for (int nc = 0; nc < 2; ++nc) {
    const __bf16* kb = K + ((long)bh * S_DIM + s0 + nc * 16) * HD;
    kf[2 * nc]     = load_b_frag(kb, HD);
    kf[2 * nc + 1] = load_b_frag(kb + 32, HD);
  }
}
__device__ __forceinline__ void load_vfrags(v16bf vf[4], const __bf16* VT, int bh, int s0) {
#pragma unroll
  for (int nc = 0; nc < 4; ++nc)
    vf[nc] = load_b_frag(VT + ((long)bh * HD + nc * 16) * S_DIM + s0, S_DIM);
}

// one 32-wide s-tile: score WMMAs, online softmax, LDS transpose, PV WMMAs
__device__ __forceinline__ void attn_step(v16bf qa0, v16bf qa1,
                                          const v16bf kf[4], const v16bf vf[4],
                                          v8f o[4], float mrow[8], float lrow[8],
                                          float* L, int ln, int hf) {
  v8f sc[2] = {};
  sc[0] = wmma_bf16(qa0, kf[0], sc[0]);
  sc[0] = wmma_bf16(qa1, kf[1], sc[0]);
  sc[1] = wmma_bf16(qa0, kf[2], sc[1]);
  sc[1] = wmma_bf16(qa1, kf[3], sc[1]);

  const float scale = 0.03125f;           // 1024^-0.5
  float mnew[8], alpha[8];
#pragma unroll
  for (int j = 0; j < 8; ++j) {
    sc[0][j] *= scale; sc[1][j] *= scale;
    float tm = row_reduce_max(fmaxf(sc[0][j], sc[1][j]));
    mnew[j] = fmaxf(mrow[j], tm);
    alpha[j] = __expf(mrow[j] - mnew[j]);
    mrow[j] = mnew[j];
  }
#pragma unroll
  for (int j = 0; j < 8; ++j) {
    sc[0][j] = __expf(sc[0][j] - mnew[j]);
    sc[1][j] = __expf(sc[1][j] - mnew[j]);
    float rs = row_reduce_sum(sc[0][j] + sc[1][j]);
    lrow[j] = lrow[j] * alpha[j] + rs;
    o[0][j] *= alpha[j]; o[1][j] *= alpha[j];
    o[2][j] *= alpha[j]; o[3][j] *= alpha[j];
    int r = j + 8 * hf;
    L[r * 32 + ln]      = sc[0][j];
    L[r * 32 + 16 + ln] = sc[1][j];
  }
  asm volatile("s_wait_dscnt 0x0" ::: "memory");
  v16bf pa;
#pragma unroll
  for (int j = 0; j < 4; ++j) {
    int k = hf * 8 + 2 * j;
    pa[2 * j]     = (__bf16)L[ln * 32 + k];
    pa[2 * j + 1] = (__bf16)L[ln * 32 + k + 1];
    int k2 = 16 + hf * 8 + 2 * j;
    pa[8 + 2 * j]     = (__bf16)L[ln * 32 + k2];
    pa[8 + 2 * j + 1] = (__bf16)L[ln * 32 + k2 + 1];
  }
#pragma unroll
  for (int nc = 0; nc < 4; ++nc)
    o[nc] = wmma_bf16(pa, vf[nc], o[nc]);
}

// grid: (B*H, T/64), block 128 (4 waves); each wave owns a 16-row q tile.
__global__ __launch_bounds__(128) void attn_kernel(const __bf16* __restrict__ Q,
                                                   const __bf16* __restrict__ K,
                                                   const __bf16* __restrict__ VT,
                                                   __bf16* __restrict__ O) {
  __shared__ float lds[4][16 * 32];
  int wave = threadIdx.x >> 5;
  int lane = threadIdx.x & 31;
  int ln = lane & 15, hf = lane >> 4;
  int bh = blockIdx.x;
  int t0 = blockIdx.y * 64 + wave * 16;

  const __bf16* qp = Q + ((long)bh * T_DIM + t0) * HD;
  v16bf qa0 = load_a_frag(qp, HD);        // K = 0..31 of hd
  v16bf qa1 = load_a_frag(qp + 32, HD);   // K = 32..63 of hd

  v8f o[4] = {};
  float mrow[8], lrow[8];
#pragma unroll
  for (int j = 0; j < 8; ++j) { mrow[j] = -3.0e38f; lrow[j] = 0.f; }
  float* L = lds[wave];

  // x2-unrolled s-loop with ping-pong K fragments: loads for one half-step
  // overlap WMMAs + softmax VALU of the other; no register copies.
  v16bf kfA[4], kfB[4], vfA[4], vfB[4];
  load_kfrags(kfA, K, bh, 0);
  for (int s0 = 0; s0 < S_DIM; s0 += 64) {
    load_vfrags(vfA, VT, bh, s0);
    load_kfrags(kfB, K, bh, s0 + 32);
    attn_step(qa0, qa1, kfA, vfA, o, mrow, lrow, L, ln, hf);
    load_vfrags(vfB, VT, bh, s0 + 32);
    if (s0 + 64 < S_DIM) load_kfrags(kfA, K, bh, s0 + 64);
    attn_step(qa0, qa1, kfB, vfB, o, mrow, lrow, L, ln, hf);
  }

  // epilogue: normalize, store concat-head layout [B, T, C]
  int b = bh >> 4, h = bh & 15;
#pragma unroll
  for (int nc = 0; nc < 4; ++nc) {
    int d = nc * 16 + ln;
#pragma unroll
    for (int j = 0; j < 8; ++j) {
      int t = t0 + j + 8 * hf;
      float val = o[nc][j] / lrow[j];
      O[((long)b * T_DIM + t) * C_DIM + h * HD + d] = (__bf16)val;
    }
  }
}

// ---------------- host launch ----------------

extern "C" void kernel_launch(void* const* d_in, const int* in_sizes, int n_in,
                              void* d_out, int out_size, void* d_ws, size_t ws_size,
                              hipStream_t stream) {
  const float* e  = (const float*)d_in[0];
  const float* x  = (const float*)d_in[1];
  const float* Wq = (const float*)d_in[2];
  const float* bq = (const float*)d_in[3];
  const float* Wk = (const float*)d_in[4];
  const float* bk = (const float*)d_in[5];
  const float* Wv = (const float*)d_in[6];
  const float* bv = (const float*)d_in[7];
  const float* Wp = (const float*)d_in[8];
  float* out = (float*)d_out;

  char* ws = (char*)d_ws;
  size_t off = 0;
  auto carve = [&](size_t bytes) -> void* {
    void* p = ws + off;
    off += (bytes + 255) & ~(size_t)255;
    return p;
  };
  const size_t ACT = (size_t)BATCH * T_DIM * C_DIM * sizeof(__bf16);   // 8 MiB
  const size_t WMT = (size_t)C_DIM * C_DIM * sizeof(__bf16);           // 2 MiB
  __bf16* xb   = (__bf16*)carve(ACT);
  __bf16* eb   = (__bf16*)carve(ACT);
  __bf16* Wqt  = (__bf16*)carve(WMT);
  __bf16* Wkt  = (__bf16*)carve(WMT);
  __bf16* Wvt  = (__bf16*)carve(WMT);
  __bf16* Wpb  = (__bf16*)carve(WMT);
  __bf16* qbuf = (__bf16*)carve(ACT);
  __bf16* kbuf = (__bf16*)carve(ACT);
  __bf16* vtbf = (__bf16*)carve(ACT);
  __bf16* attb = (__bf16*)carve(ACT);

  const int NACT = BATCH * T_DIM * C_DIM;   // 4M
  const int NW   = C_DIM * C_DIM;           // 1M

  cvt_f32_bf16_kernel<<<(NACT + 255) / 256, 256, 0, stream>>>(x, xb, NACT);
  cvt_f32_bf16_kernel<<<(NACT + 255) / 256, 256, 0, stream>>>(e, eb, NACT);
  cvt_f32_bf16_kernel<<<(NW + 255) / 256, 256, 0, stream>>>(Wp, Wpb, NW);
  wqkv_transpose_kernel<<<(NW + 255) / 256, 256, 0, stream>>>(Wq, Wqt);
  wqkv_transpose_kernel<<<(NW + 255) / 256, 256, 0, stream>>>(Wk, Wkt);
  wqkv_transpose_kernel<<<(NW + 255) / 256, 256, 0, stream>>>(Wv, Wvt);

  gemm_qkv_kernel<<<512, 256, 0, stream>>>(xb, Wqt, bq, qbuf, 0);
  gemm_qkv_kernel<<<512, 256, 0, stream>>>(eb, Wkt, bk, kbuf, 1);
  gemm_qkv_kernel<<<512, 256, 0, stream>>>(eb, Wvt, bv, vtbf, 2);

  dim3 ag(BATCH * NHEAD, T_DIM / 64);
  attn_kernel<<<ag, 128, 0, stream>>>(qbuf, kbuf, vtbf, attb);

  gemm_out_kernel<<<512, 256, 0, stream>>>(attb, Wpb, out);
}
